// InferCapsule_83013127897546
// MI455X (gfx1250) — matrix operand order repaired and checked by reading, not compiled, not run on hardware
//
#include <hip/hip_runtime.h>
#include <math.h>

#define B_ 256
#define M_ 64
#define N_ 32
#define D_ 128
#define T_ 64
#define L_ 128

typedef __attribute__((ext_vector_type(16))) __bf16 v16bf;
typedef __attribute__((ext_vector_type(8)))  __bf16 v8bf;
typedef __attribute__((ext_vector_type(8)))  float  v8f;

// ---------------------------------------------------------------------------
// Kernel 1: exact 1.5-entmax over the D axis of route_weights [M,N,D,T].
// One block per (m,n); LDS-staged [D][T] slab; one wave per 8 t-columns.
// tau* found by bisection on f(tau) = sum_d max(x_d - tau, 0)^2 = 1,
// which is the unique solution the reference's sort algorithm computes.
// ---------------------------------------------------------------------------
__global__ __launch_bounds__(256)
void entmax15_kernel(const float* __restrict__ rw, __bf16* __restrict__ wbf) {
    __shared__ float slab[D_][T_ + 1];
    const int mn   = blockIdx.x;        // 0..M*N-1
    const int tid  = threadIdx.x;
    const int lane = tid & 31;
    const int wv   = tid >> 5;
    const size_t base = (size_t)mn * (D_ * T_);

    for (int idx = tid; idx < D_ * T_; idx += 256)
        slab[idx >> 6][idx & 63] = rw[base + idx];
    __syncthreads();

    for (int i = 0; i < 8; ++i) {
        const int t = wv * 8 + i;
        float x0 = slab[lane      ][t] * 0.5f;
        float x1 = slab[lane + 32 ][t] * 0.5f;
        float x2 = slab[lane + 64 ][t] * 0.5f;
        float x3 = slab[lane + 96 ][t] * 0.5f;
        float mx = fmaxf(fmaxf(x0, x1), fmaxf(x2, x3));
        for (int off = 16; off > 0; off >>= 1)
            mx = fmaxf(mx, __shfl_xor(mx, off, 32));
        x0 -= mx; x1 -= mx; x2 -= mx; x3 -= mx;

        float lo = -1.0f, hi = 0.0f;
        for (int it = 0; it < 32; ++it) {
            float tau = 0.5f * (lo + hi);
            float a0 = fmaxf(x0 - tau, 0.0f), a1 = fmaxf(x1 - tau, 0.0f);
            float a2 = fmaxf(x2 - tau, 0.0f), a3 = fmaxf(x3 - tau, 0.0f);
            float s = a0*a0 + a1*a1 + a2*a2 + a3*a3;
            for (int off = 16; off > 0; off >>= 1)
                s += __shfl_xor(s, off, 32);
            if (s > 1.0f) lo = tau; else hi = tau;
        }
        const float tau = 0.5f * (lo + hi);
        float p0 = fmaxf(x0 - tau, 0.0f); p0 *= p0;
        float p1 = fmaxf(x1 - tau, 0.0f); p1 *= p1;
        float p2 = fmaxf(x2 - tau, 0.0f); p2 *= p2;
        float p3 = fmaxf(x3 - tau, 0.0f); p3 *= p3;
        slab[lane      ][t] = p0;
        slab[lane + 32 ][t] = p1;
        slab[lane + 64 ][t] = p2;
        slab[lane + 96 ][t] = p3;
    }
    __syncthreads();
    for (int idx = tid; idx < D_ * T_; idx += 256)
        wbf[base + idx] = (__bf16)slab[idx >> 6][idx & 63];
}

// ---------------------------------------------------------------------------
// Kernel 2: normalized leave hash in natural [L,T] layout, bf16.
// ---------------------------------------------------------------------------
__global__ void lhash_kernel(const float* __restrict__ rl, __bf16* __restrict__ lh) {
    const int l = threadIdx.x;          // 128 threads
    float s = 0.0f;
    for (int t = 0; t < T_; ++t) { float v = rl[l * T_ + t]; s += v * v; }
    const float inv = 1.0f / fmaxf(sqrtf(s), 1e-12f);
    for (int t = 0; t < T_; ++t)
        lh[l * T_ + t] = (__bf16)(rl[l * T_ + t] * inv);
}

// A fragment (16-bit A 16x32 layout): two contiguous 8-element runs per lane.
__device__ __forceinline__ v16bf make_afrag(const __bf16* p) {
    const v8bf lo = *(const v8bf*)(p);          // K = kb+0 .. kb+7
    const v8bf hi = *(const v8bf*)(p + 16);     // K = kb+16 .. kb+23
    return __builtin_shufflevector(lo, hi, 0, 1, 2, 3, 4, 5, 6, 7,
                                           8, 9, 10, 11, 12, 13, 14, 15);
}

// ---------------------------------------------------------------------------
// Kernel 3: priors[b,m,n,t] = sum_d x[b,m,d] * weights[m,n,d,t].
// One block per (m,n): GEMM 256(b) x 64(t) x 128(d), bf16 WMMA, f32 accum.
// W staged TRANSPOSED in LDS ([t][d]) so each B fragment is two contiguous
// 16B reads (2x ds_load_b128) instead of 16 ds_load_u16.
// ---------------------------------------------------------------------------
__global__ __launch_bounds__(256)
void priors_gemm_kernel(const float* __restrict__ x, const __bf16* __restrict__ wbf,
                        float* __restrict__ priors) {
    __shared__ alignas(16) __bf16 Xs[128][136];   // [b_local][d], 272B stride
    __shared__ alignas(16) __bf16 Wt[T_][136];    // [t][d],       272B stride

    const int mn   = blockIdx.x;
    const int m    = mn >> 5;           // / N_
    const int n    = mn & 31;
    const int tid  = threadIdx.x;
    const int lane = tid & 31;
    const int wv   = tid >> 5;

    {   // W tile: contiguous 8192 bf16 in global, transposed into LDS
        const size_t wbase = (size_t)mn * (D_ * T_);
        for (int idx = tid; idx < D_ * T_; idx += 256)
            Wt[idx & 63][idx >> 6] = wbf[wbase + idx];
    }

    const int row  = lane & 15;
    const int hsel = lane >> 4;         // half-wave select
    const int akb  = hsel * 8;          // A-frag K base (16-bit A 16x32 layout)
    const int bkb  = hsel * 8;          // B-frag K base (same K order as A)

    for (int half = 0; half < 2; ++half) {
        __syncthreads();                // Wt visible / previous half's compute done
        for (int idx = tid; idx < 128 * D_; idx += 256) {
            const int bl = idx >> 7;
            const int d  = idx & 127;
            const int bg = half * 128 + bl;
            Xs[bl][d] = (__bf16)x[((size_t)bg * M_ + m) * D_ + d];
        }
        __syncthreads();

        // wave wv: b-tile rows [wv*16, wv*16+16) x all 4 t-tiles, K = 128
        v8f acc[4] = {};
        const int b0 = wv * 16;
        #pragma unroll
        for (int ks = 0; ks < 4; ++ks) {
            const int k0 = ks * 32;
            const v16bf a = make_afrag(&Xs[b0 + row][k0 + akb]);
            #pragma unroll
            for (int tt = 0; tt < 4; ++tt) {
                const v16bf bf = make_afrag(&Wt[tt * 16 + row][k0 + bkb]);
                acc[tt] = __builtin_amdgcn_wmma_f32_16x16x32_bf16(
                    false, a, false, bf, (short)0, acc[tt], false, false);
            }
        }

        #pragma unroll
        for (int tt = 0; tt < 4; ++tt) {
            const int t = tt * 16 + row;
            #pragma unroll
            for (int r = 0; r < 8; ++r) {
                const int bg = half * 128 + b0 + r + hsel * 8;
                priors[(size_t)bg * (M_ * N_ * T_) + (size_t)m * (N_ * T_)
                       + (size_t)n * T_ + t] = acc[tt][r];
            }
        }
    }
}

// ---------------------------------------------------------------------------
// Kernel 4: fully fused routing per (b,n):
//   votes = sigmoid(priors @ leave_hash^T)  (WMMA, 64x128)
//   dis   = mean_l (votes - mean_m votes)^2
//   prob  = softmax_m relu(thread^2 - dis)
//   out   = LayerNorm_t( sum_m prob * priors )
// Leave hash staged as [l][t] so B fragments are two contiguous 16B reads.
// ---------------------------------------------------------------------------
__global__ __launch_bounds__(256)
void routing_kernel(const float* __restrict__ priors, const __bf16* __restrict__ lh,
                    const float* __restrict__ thread_in,
                    const float* __restrict__ gamma, const float* __restrict__ beta,
                    float* __restrict__ out) {
    __shared__ alignas(16) __bf16 Pb[M_][72];     // priors bf16 [m][t], 144B stride
    __shared__ alignas(16) __bf16 Lht[L_][72];    // leave hash [l][t], 144B stride
    __shared__ float  votes[M_][132];             // [m][l]
    __shared__ float  meanc[L_];
    __shared__ float  warr[M_];
    __shared__ float  parr[M_];
    __shared__ float  ncap[T_];

    const int blk  = blockIdx.x;
    const int b    = blk >> 5;          // / N_
    const int n    = blk & 31;
    const int tid  = threadIdx.x;
    const int lane = tid & 31;
    const int wv   = tid >> 5;
    const size_t pbase = (size_t)b * (M_ * N_ * T_) + (size_t)n * T_;

    for (int idx = tid; idx < M_ * T_; idx += 256) {
        const int m = idx >> 6, t = idx & 63;
        Pb[m][t] = (__bf16)priors[pbase + (size_t)m * (N_ * T_) + t];
    }
    for (int idx = tid; idx < L_ * T_; idx += 256)
        Lht[idx >> 6][idx & 63] = lh[idx];
    __syncthreads();

    {   // votes GEMM: M=64(m) x N=128(l) x K=64(t); 4 C-tiles per wave
        const int mt   = wv & 3;
        const int ltb  = (wv >> 2) * 4;
        const int row  = lane & 15;
        const int hsel = lane >> 4;
        const int akb  = hsel * 8;
        const int bkb  = hsel * 8;
        v8f acc[4] = {};
        #pragma unroll
        for (int ks = 0; ks < 2; ++ks) {
            const int k0 = ks * 32;
            const v16bf a = make_afrag(&Pb[mt * 16 + row][k0 + akb]);
            #pragma unroll
            for (int i = 0; i < 4; ++i) {
                const int l0 = (ltb + i) * 16;
                const v16bf bf = make_afrag(&Lht[l0 + row][k0 + bkb]);
                acc[i] = __builtin_amdgcn_wmma_f32_16x16x32_bf16(
                    false, a, false, bf, (short)0, acc[i], false, false);
            }
        }
        #pragma unroll
        for (int i = 0; i < 4; ++i) {
            const int l0 = (ltb + i) * 16;
            #pragma unroll
            for (int r = 0; r < 8; ++r) {
                const int mm = mt * 16 + r + hsel * 8;
                votes[mm][l0 + row] = 1.0f / (1.0f + __expf(-acc[i][r]));
            }
        }
    }
    __syncthreads();

    if (tid < L_) {                     // mean over m
        float s = 0.0f;
        for (int m = 0; m < M_; ++m) s += votes[m][tid];
        meanc[tid] = s * (1.0f / M_);
    }
    __syncthreads();

    if (tid < M_) {                     // dis + relu(thread^2 - dis)
        float s = 0.0f;
        for (int l = 0; l < L_; ++l) {
            const float d = votes[tid][l] - meanc[l];
            s += d * d;
        }
        const float dis = s * (1.0f / L_);
        const float th  = thread_in[tid * N_ + n];
        warr[tid] = fmaxf(th * th - dis, 0.0f);
    }
    __syncthreads();

    if (tid < M_) {                     // softmax over m
        float mx = -1e30f;
        for (int m = 0; m < M_; ++m) mx = fmaxf(mx, warr[m]);
        float se = 0.0f;
        for (int m = 0; m < M_; ++m) se += __expf(warr[m] - mx);
        parr[tid] = __expf(warr[tid] - mx) / se;
    }
    __syncthreads();

    if (tid < T_) {                     // next_caps (fp32 priors for accuracy)
        float s = 0.0f;
        for (int m = 0; m < M_; ++m)
            s += parr[m] * priors[pbase + (size_t)m * (N_ * T_) + tid];
        ncap[tid] = s;
    }
    __syncthreads();

    if (tid < T_) {                     // LayerNorm over t
        float mu = 0.0f;
        for (int t = 0; t < T_; ++t) mu += ncap[t];
        mu *= (1.0f / T_);
        float var = 0.0f;
        for (int t = 0; t < T_; ++t) { const float d = ncap[t] - mu; var += d * d; }
        var *= (1.0f / T_);
        const float y = (ncap[tid] - mu) * rsqrtf(var + 1e-5f) * gamma[tid] + beta[tid];
        out[(size_t)(b * N_ + n) * T_ + tid] = y;
    }
}

// ---------------------------------------------------------------------------
extern "C" void kernel_launch(void* const* d_in, const int* in_sizes, int n_in,
                              void* d_out, int out_size, void* d_ws, size_t ws_size,
                              hipStream_t stream) {
    (void)in_sizes; (void)n_in; (void)out_size; (void)ws_size;
    const float* x  = (const float*)d_in[0];   // [B,M,D]
    const float* rw = (const float*)d_in[1];   // [M,N,D,T]
    const float* th = (const float*)d_in[2];   // [1,M,N]
    const float* rl = (const float*)d_in[3];   // [L,T]
    const float* g  = (const float*)d_in[4];   // [T]
    const float* bt = (const float*)d_in[5];   // [T]
    float* out = (float*)d_out;                // [B,N,T]

    char* ws = (char*)d_ws;
    __bf16* wbf    = (__bf16*)ws;                                      // 32 MB
    __bf16* lhash  = (__bf16*)(ws + (size_t)(32u << 20));              // 16 KB
    float*  priors = (float*)(ws + (size_t)(32u << 20) + (64u << 10)); // 128 MB

    entmax15_kernel   <<<M_ * N_, 256, 0, stream>>>(rw, wbf);
    lhash_kernel      <<<1, L_, 0, stream>>>(rl, lhash);
    priors_gemm_kernel<<<M_ * N_, 256, 0, stream>>>(x, wbf, priors);
    routing_kernel    <<<B_ * N_, 256, 0, stream>>>(priors, lhash, th, g, bt, out);
}
